// AllGraphBlock_50182397886932
// MI455X (gfx1250) — matrix operand order repaired and checked by reading, not compiled, not run on hardware
//
#include <hip/hip_runtime.h>
#include <math.h>

// ============================================================================
// MoNet / GMMConv GNN forward for MI455X (gfx1250, wave32).
// Dense x@g transforms: LDS-staged 32x64 tiles, 2x V_WMMA_F32_16X16X32_F16
// per K-chunk per wave (shared B fragment). Edge message passing: per-edge
// Gaussian weights + float4 atomic scatter-add. BN: one block per channel.
// Pools: ordered-uint atomicMax.
// ============================================================================

typedef __attribute__((ext_vector_type(16))) _Float16 v16h;
typedef __attribute__((ext_vector_type(8)))  _Float16 v8h;
typedef __attribute__((ext_vector_type(8)))  float    v8f;

#define G_GRAPHS 40
#define CDIVL(a,b) (unsigned)(((long long)(a) + (b) - 1) / (b))

static __device__ __forceinline__ float eluf(float v) {
  return v > 0.0f ? v : (__expf(v) - 1.0f);
}

// ---------------------------------------------------------------------------
// C[n,cols] = A[n,kdim] * B[kdim,cols]; fp32 storage, f16 WMMA, f32 accum.
// Block = 128 threads (4 waves). Block tile: 32 rows x 64 cols; wave w owns
// a 16-col strip and computes two row-halves (two accumulators sharing the
// same B fragment -> 2 independent WMMAs per K-chunk). K staged via LDS in
// chunks of 32. Staging: batch loads first (single wait), vector ds stores;
// full in-range chunks use unconditional float4 global loads.
// Requires cols % 8 == 0 (true for all calls here).
// ---------------------------------------------------------------------------
__global__ __launch_bounds__(128) void k_wmma_gemm(
    const float* __restrict__ A, const float* __restrict__ B,
    float* __restrict__ C, int n, int kdim, int cols)
{
  __shared__ _Float16 aLds[32 * 32];      // [row][k], stride 32 halves
  __shared__ _Float16 bLds[64 * 40];      // [col][k], stride 40 halves (pad 8)

  const int tid  = threadIdx.y * 32 + threadIdx.x;
  const int lane = threadIdx.x;
  const int lm   = lane & 15;
  const int hi   = lane >> 4;
  const int w    = threadIdx.y;           // wave id -> col sub-tile
  const int colBase = blockIdx.x * 64;
  const int row0    = blockIdx.y * 32;
  const int bCol    = colBase + w * 16 + lm;
  const bool cOk    = bCol < cols;

  // A staging map: thread -> (row ar 0..31, 8 consecutive k at ak)
  const int ar = tid >> 2;
  const int ak = (tid & 3) * 8;
  const int   aRowG = row0 + ar;
  const int   aRowC = aRowG < n ? aRowG : (n - 1);
  const float aRm   = aRowG < n ? 1.0f : 0.0f;
  const float* aRowP = A + (size_t)aRowC * kdim;
  _Float16* aSlot = aLds + ar * 32 + ak;  // 16B-aligned (ak % 8 == 0)

  // B staging map: thread -> fixed col quad, k rows brBase + {0,8,16,24}
  const int   bc4    = (tid & 15) * 4;    // col within block tile (invariant)
  const int   brBase = tid >> 4;          // 0..7 (invariant)
  const int   c4     = colBase + bc4;
  const bool  c4ok   = (c4 + 3) < cols;   // whole quad valid (cols % 8 == 0)
  const int   c4c    = c4ok ? c4 : (cols - 4);
  const float cmask  = c4ok ? 1.0f : 0.0f;

  const bool aVecOk = (kdim & 3) == 0;    // float4-aligned A rows

  v8f acc0 = {}, acc1 = {};
  for (int k0 = 0; k0 < kdim; k0 += 32) {
    // ---- stage A chunk (32 x 32) -> f16 LDS (one ds_store_b128) ----
    v8h ah;
    if (aVecOk && (k0 + 32 <= kdim)) {
      // fast path: whole chunk in range, two unconditional b128 loads
      const float4 va0 = *(const float4*)(aRowP + k0 + ak);
      const float4 va1 = *(const float4*)(aRowP + k0 + ak + 4);
      ah[0] = (_Float16)(va0.x * aRm); ah[1] = (_Float16)(va0.y * aRm);
      ah[2] = (_Float16)(va0.z * aRm); ah[3] = (_Float16)(va0.w * aRm);
      ah[4] = (_Float16)(va1.x * aRm); ah[5] = (_Float16)(va1.y * aRm);
      ah[6] = (_Float16)(va1.z * aRm); ah[7] = (_Float16)(va1.w * aRm);
    } else {
      // tail path: clamped loads batched first (single wait), mask after
      float av[8];
#pragma unroll
      for (int j = 0; j < 8; ++j) {
        const int kk = k0 + ak + j;
        av[j] = aRowP[kk < kdim ? kk : (kdim - 1)];
      }
#pragma unroll
      for (int j = 0; j < 8; ++j) {
        const float m = ((k0 + ak + j) < kdim) ? aRm : 0.0f;
        ah[j] = (_Float16)(av[j] * m);
      }
    }
    *(v8h*)aSlot = ah;

    // ---- stage B chunk (32 x 64) transposed -> bLds[col][k] ----
    {
      float4 vb[4];
#pragma unroll
      for (int j = 0; j < 4; ++j) {
        const int kk  = k0 + brBase + j * 8;
        const int kkc = kk < kdim ? kk : (kdim - 1);
        vb[j] = *(const float4*)(B + (size_t)kkc * cols + c4c);
      }
#pragma unroll
      for (int j = 0; j < 4; ++j) {
        const int   br = brBase + j * 8;
        const float mm = ((k0 + br) < kdim) ? cmask : 0.0f;
        bLds[(bc4 + 0) * 40 + br] = (_Float16)(vb[j].x * mm);
        bLds[(bc4 + 1) * 40 + br] = (_Float16)(vb[j].y * mm);
        bLds[(bc4 + 2) * 40 + br] = (_Float16)(vb[j].z * mm);
        bLds[(bc4 + 3) * 40 + br] = (_Float16)(vb[j].w * mm);
      }
    }
    if (k0 + 32 < kdim) {                  // uniform-condition prefetch
      __builtin_prefetch(B + (size_t)(k0 + 32 + brBase) * cols + c4c, 0, 1);
      __builtin_prefetch(aRowP + k0 + 32 + ak, 0, 1);
    }
    __syncthreads();

    // ---- pack WMMA operands from LDS (contiguous 16B v8h reads) ----
    // A f16 16x32 layout: h<8 -> k = hi*8 + h ; h>=8 -> k = 16 + hi*8 + (h-8)
    const v8h* aV = (const v8h*)aLds;      // 4 v8h per row
    const v8h a0lo = aV[lm * 4 + hi];
    const v8h a0hi = aV[lm * 4 + 2 + hi];
    const v8h a1lo = aV[(16 + lm) * 4 + hi];
    const v8h a1hi = aV[(16 + lm) * 4 + 2 + hi];
    const v16h a0 = __builtin_shufflevector(a0lo, a0hi,
        0,1,2,3,4,5,6,7,8,9,10,11,12,13,14,15);
    const v16h a1 = __builtin_shufflevector(a1lo, a1hi,
        0,1,2,3,4,5,6,7,8,9,10,11,12,13,14,15);
    // B f16 32x16 layout: lane col = w*16+lm, h<8 -> k = hi*8 + h ; h>=8 -> k = 16 + hi*8 + (h-8)
    const v8h* bV = (const v8h*)bLds;      // 5 v8h per col (stride 40 halves)
    const int  bb = (w * 16 + lm) * 5 + hi;
    const v16h b = __builtin_shufflevector(bV[bb], bV[bb + 2],
        0,1,2,3,4,5,6,7,8,9,10,11,12,13,14,15);

    acc0 = __builtin_amdgcn_wmma_f32_16x16x32_f16(
        false, a0, false, b, (short)0, acc0, false, false);
    acc1 = __builtin_amdgcn_wmma_f32_16x16x32_f16(
        false, a1, false, b, (short)0, acc1, false, false);
    __syncthreads();
  }
  // D layout: VGPR r -> M = r (lanes 0-15) / 8+r (lanes 16-31), N = lane%16.
#pragma unroll
  for (int r = 0; r < 8; ++r) {
    const int rowA = row0 + r + (hi ? 8 : 0);
    const int rowB = rowA + 16;
    if (rowA < n && cOk) C[(size_t)rowA * cols + bCol] = acc0[r];
    if (rowB < n && cOk) C[(size_t)rowB * cols + bCol] = acc1[r];
  }
}

// ---------------------------------------------------------------------------
// Gaussian mixture weights per edge: gw[e,k] = exp(-0.5*sum(d^2/(s^2+eps)))
// ---------------------------------------------------------------------------
__global__ void k_edge_gw(const float* __restrict__ ea,
                          const float* __restrict__ mu,
                          const float* __restrict__ sigma,
                          float* __restrict__ gw, int E, int K)
{
  int e = blockIdx.x * blockDim.x + threadIdx.x;
  if (e >= E) return;
  float ex = ea[3*e], ey = ea[3*e+1], ez = ea[3*e+2];
  for (int k = 0; k < K; ++k) {
    float dx = ex - mu[3*k], dy = ey - mu[3*k+1], dz = ez - mu[3*k+2];
    float sx = sigma[3*k], sy = sigma[3*k+1], sz = sigma[3*k+2];
    float q = dx*dx/(sx*sx+1e-15f) + dy*dy/(sy*sy+1e-15f) + dz*dz/(sz*sz+1e-15f);
    gw[(size_t)e*K + k] = __expf(-0.5f*q);
  }
}

__global__ void k_edge_deg(const int* __restrict__ dst, float* __restrict__ deg, int E)
{
  int e = blockIdx.x * blockDim.x + threadIdx.x;
  if (e >= E) return;
  __hip_atomic_fetch_add(&deg[dst[e]], 1.0f, __ATOMIC_RELAXED, __HIP_MEMORY_SCOPE_AGENT);
}

// msg[e, 4m..4m+3] = sum_k gw[e,k] * xg[src[e], k, 4m..4m+3]; scatter into agg.
// Requires M % 4 == 0 (true: all couts are multiples of 8).
__global__ void k_edge_msg(const float* __restrict__ xg, const float* __restrict__ gw,
                           const int* __restrict__ src, const int* __restrict__ dst,
                           float* __restrict__ agg, int E, int K, int M4, int M)
{
  long long idx = (long long)blockIdx.x * blockDim.x + threadIdx.x;
  if (idx >= (long long)E * M4) return;
  const int e  = (int)(idx / M4), m4 = (int)(idx % M4);
  const float4* xr = (const float4*)(xg + (size_t)src[e] * K * M) + m4;
  const float*  gr = gw + (size_t)e * K;
  float4 a = {0.f, 0.f, 0.f, 0.f};
  for (int k = 0; k < K; ++k) {
    const float  g = gr[k];
    const float4 v = xr[(size_t)k * M4];
    a.x += g * v.x; a.y += g * v.y; a.z += g * v.z; a.w += g * v.w;
  }
  float* base = agg + (size_t)dst[e] * M + m4 * 4;
  __hip_atomic_fetch_add(base + 0, a.x, __ATOMIC_RELAXED, __HIP_MEMORY_SCOPE_AGENT);
  __hip_atomic_fetch_add(base + 1, a.y, __ATOMIC_RELAXED, __HIP_MEMORY_SCOPE_AGENT);
  __hip_atomic_fetch_add(base + 2, a.z, __ATOMIC_RELAXED, __HIP_MEMORY_SCOPE_AGENT);
  __hip_atomic_fetch_add(base + 3, a.w, __ATOMIC_RELAXED, __HIP_MEMORY_SCOPE_AGENT);
}

// out = agg/max(deg,1) + rootout + bias
__global__ void k_finalize(const float* __restrict__ agg, const float* __restrict__ deg,
                           const float* __restrict__ rootout, const float* __restrict__ bias,
                           float* __restrict__ out, int n, int M)
{
  long long idx = (long long)blockIdx.x * blockDim.x + threadIdx.x;
  if (idx >= (long long)n * M) return;
  int i = (int)(idx / M), m = (int)(idx % M);
  float d = deg[i]; d = d > 1.0f ? d : 1.0f;
  out[idx] = agg[idx] / d + rootout[idx] + bias[m];
}

// ---------------------------------------------------------------------------
// BatchNorm: one block per channel -> stats[ch]=mean, stats[C+ch]=var (biased)
// ---------------------------------------------------------------------------
__global__ __launch_bounds__(256) void k_bn_stats(const float* __restrict__ x,
                                                  int n, int C, float* __restrict__ stats)
{
  __shared__ float ssum[256], ssq[256];
  int ch = blockIdx.x;
  float s = 0.0f, q = 0.0f;
  for (int i = threadIdx.x; i < n; i += 256) {
    float v = x[(size_t)i * C + ch];
    s += v; q += v * v;
  }
  ssum[threadIdx.x] = s; ssq[threadIdx.x] = q;
  __syncthreads();
  for (int w = 128; w > 0; w >>= 1) {
    if (threadIdx.x < w) {
      ssum[threadIdx.x] += ssum[threadIdx.x + w];
      ssq[threadIdx.x]  += ssq[threadIdx.x + w];
    }
    __syncthreads();
  }
  if (threadIdx.x == 0) {
    float mean = ssum[0] / (float)n;
    stats[ch]     = mean;
    stats[C + ch] = ssq[0] / (float)n - mean * mean;
  }
}

// mode 0: bn; mode 1: elu(bn); mode 2: elu(bn + skip)
__global__ void k_bn_apply(const float* __restrict__ in, const float* __restrict__ stats,
                           const float* __restrict__ gamma, const float* __restrict__ beta,
                           const float* __restrict__ skip, float* __restrict__ out,
                           int n, int C, int mode)
{
  long long idx = (long long)blockIdx.x * blockDim.x + threadIdx.x;
  if (idx >= (long long)n * C) return;
  int ch = (int)(idx % C);
  float y = (in[idx] - stats[ch]) * rsqrtf(stats[C + ch] + 1e-5f) * gamma[ch] + beta[ch];
  if (mode == 1)      y = eluf(y);
  else if (mode == 2) y = eluf(y + skip[idx]);
  out[idx] = y;
}

// ---------------------------------------------------------------------------
// Segment max pooling via order-preserving uint encoding + atomicMax
// ---------------------------------------------------------------------------
static __device__ __forceinline__ unsigned fenc(float f) {
  unsigned u = __float_as_uint(f);
  return (u & 0x80000000u) ? ~u : (u | 0x80000000u);
}
static __device__ __forceinline__ float fdec(unsigned u) {
  return __uint_as_float((u & 0x80000000u) ? (u & 0x7FFFFFFFu) : ~u);
}

__global__ void k_pool_scatter(const float* __restrict__ x, const int* __restrict__ cluster,
                               unsigned* __restrict__ umax, int n_in, int C)
{
  long long idx = (long long)blockIdx.x * blockDim.x + threadIdx.x;
  if (idx >= (long long)n_in * C) return;
  int i = (int)(idx / C), ch = (int)(idx % C);
  atomicMax(&umax[(size_t)cluster[i] * C + ch], fenc(x[idx]));
}

__global__ void k_pool_decode(const unsigned* __restrict__ umax, float* __restrict__ out,
                              int n_out, int C)
{
  long long idx = (long long)blockIdx.x * blockDim.x + threadIdx.x;
  if (idx >= (long long)n_out * C) return;
  out[idx] = fdec(umax[idx]);
}

// ---------------------------------------------------------------------------
// Global mean pool over graphs
// ---------------------------------------------------------------------------
__global__ void k_graph_sum(const float* __restrict__ x, const int* __restrict__ batch,
                            float* __restrict__ gsum, float* __restrict__ gcnt, int n, int C)
{
  long long idx = (long long)blockIdx.x * blockDim.x + threadIdx.x;
  if (idx >= (long long)n * C) return;
  int i = (int)(idx / C), ch = (int)(idx % C);
  int b = batch[i];
  __hip_atomic_fetch_add(&gsum[(size_t)b * C + ch], x[idx],
                         __ATOMIC_RELAXED, __HIP_MEMORY_SCOPE_AGENT);
  if (ch == 0)
    __hip_atomic_fetch_add(&gcnt[b], 1.0f, __ATOMIC_RELAXED, __HIP_MEMORY_SCOPE_AGENT);
}

__global__ void k_graph_mean(const float* __restrict__ gsum, const float* __restrict__ gcnt,
                             float* __restrict__ out, int C)
{
  long long idx = (long long)blockIdx.x * blockDim.x + threadIdx.x;
  if (idx >= (long long)G_GRAPHS * C) return;
  int g = (int)(idx / C);
  float c = gcnt[g]; c = c > 1.0f ? c : 1.0f;
  out[idx] = gsum[idx] / c;
}

// ===========================================================================
// Host orchestration.
// Input index map (setup_inputs insertion order; nested 'params' dict also
// flattened in insertion order):
//   0 x | 1-4 ea0..ea3
//   5-9 conv1{g,mu,sigma,root,bias} | 10-11 bn1{gamma,beta}
//   12-32 b1{lc1(5),lbn1(2),lc2(5),lbn2(2),sc(5),sbn(2)} | 33-53 b2 | 54-74 b3
//   75-78 ei0..ei3 | 79-82 c1..c4 | 83 batch4 | 84-87 n1..n4 (unused; sizes
//   derived from in_sizes: n_l = len(c_{l+1}), n4 = len(batch4))
// ===========================================================================
extern "C" void kernel_launch(void* const* d_in, const int* in_sizes, int n_in,
                              void* d_out, int out_size, void* d_ws, size_t ws_size,
                              hipStream_t stream)
{
  auto F  = [&](int i) { return (const float*)d_in[i]; };
  auto Iv = [&](int i) { return (const int*)d_in[i]; };

  struct Gmm { const float *g, *mu, *sigma, *root, *bias; int K; };
  struct Bn  { const float *gamma, *beta; };
  struct Res { Gmm lc1; Bn lbn1; Gmm lc2; Bn lbn2; Gmm sc; Bn sbn; };
  auto gmmAt = [&](int b, int K) { return Gmm{F(b), F(b+1), F(b+2), F(b+3), F(b+4), K}; };
  auto bnAt  = [&](int b)        { return Bn{F(b), F(b+1)}; };
  auto resAt = [&](int b) {
    return Res{gmmAt(b,5), bnAt(b+5), gmmAt(b+7,5), bnAt(b+12), gmmAt(b+14,1), bnAt(b+19)};
  };

  const float* x      = F(0);
  const float* ea[4]  = {F(1), F(2), F(3), F(4)};
  Gmm conv1 = gmmAt(5, 5);  Bn bn1 = bnAt(10);
  Res RB[3] = {resAt(12), resAt(33), resAt(54)};
  const int* ei[4]    = {Iv(75), Iv(76), Iv(77), Iv(78)};
  const int* cl[4]    = {Iv(79), Iv(80), Iv(81), Iv(82)};
  const int* batch4   = Iv(83);

  const int N    = in_sizes[0] / 3;
  const int E[4] = {in_sizes[1]/3, in_sizes[2]/3, in_sizes[3]/3, in_sizes[4]/3};
  const int n1 = in_sizes[80], n2 = in_sizes[81], n3 = in_sizes[82], n4 = in_sizes[83];
  const int nodes[5] = {N, n1, n2, n3, n4};
  const int coutL[4] = {24, 48, 96, 192};

  // ---- workspace bump allocator ----
  size_t off = 0;
  auto alloc = [&](size_t bytes) -> void* {
    void* p = (char*)d_ws + off;
    off += (bytes + 255) & ~(size_t)255;
    return p;
  };
  auto fmax2 = [](size_t a, size_t b) { return a > b ? a : b; };

  size_t maxNC = 0, maxXG = 0, maxBlk = 0, maxPool = 0;
  for (int l = 0; l < 4; ++l) {
    maxNC = fmax2(maxNC, (size_t)nodes[l] * coutL[l]);
    maxXG = fmax2(maxXG, (size_t)nodes[l] * 5 * coutL[l]);
  }
  maxBlk  = fmax2(fmax2((size_t)n1*48, (size_t)n2*96), (size_t)n3*192);
  maxPool = fmax2(fmax2((size_t)n1*24, (size_t)n2*48),
                  fmax2((size_t)n3*96, (size_t)n4*192));
  size_t maxE = 0; for (int l = 0; l < 4; ++l) maxE = fmax2(maxE, (size_t)E[l]);

  float* h0   = (float*)alloc((size_t)N  * 24  * 4);
  float* p1   = (float*)alloc((size_t)n1 * 24  * 4);
  float* b1o  = (float*)alloc((size_t)n1 * 48  * 4);
  float* p2   = (float*)alloc((size_t)n2 * 48  * 4);
  float* b2o  = (float*)alloc((size_t)n2 * 96  * 4);
  float* p3   = (float*)alloc((size_t)n3 * 96  * 4);
  float* b3o  = (float*)alloc((size_t)n3 * 192 * 4);
  float* p4   = (float*)alloc((size_t)n4 * 192 * 4);
  float* degB[4];
  for (int l = 0; l < 4; ++l) degB[l] = (float*)alloc((size_t)nodes[l] * 4);
  float* xg      = (float*)alloc(maxXG * 4);
  float* rootout = (float*)alloc(maxNC * 4);
  float* agg     = (float*)alloc(maxNC * 4);
  float* gw      = (float*)alloc(maxE * 5 * 4);
  float* th1     = (float*)alloc(maxBlk * 4);
  float* th2     = (float*)alloc(maxBlk * 4);
  float* ts      = (float*)alloc(maxBlk * 4);
  unsigned* umax = (unsigned*)alloc(maxPool * 4);
  float* stats   = (float*)alloc(2 * 192 * 4);
  float* gsum    = (float*)alloc((size_t)G_GRAPHS * 192 * 4);
  float* gcnt    = (float*)alloc((size_t)G_GRAPHS * 4);

  // ---- helpers ----
  auto gemm = [&](const float* A, const float* B, float* Cb, int nn, int kk, int cc) {
    dim3 grid(CDIVL(cc, 64), CDIVL(nn, 32));
    k_wmma_gemm<<<grid, dim3(32, 4), 0, stream>>>(A, B, Cb, nn, kk, cc);
  };
  auto runConv = [&](const float* xin, int n, int cin, int cout, const Gmm& p,
                     const int* eis, int Ecnt, const float* eaf,
                     const float* degv, float* out) {
    gemm(xin, p.g,    xg,      n, cin, p.K * cout);
    gemm(xin, p.root, rootout, n, cin, cout);
    hipMemsetAsync(agg, 0, (size_t)n * cout * 4, stream);
    k_edge_gw<<<CDIVL(Ecnt, 256), 256, 0, stream>>>(eaf, p.mu, p.sigma, gw, Ecnt, p.K);
    k_edge_msg<<<CDIVL((long long)Ecnt * (cout/4), 256), 256, 0, stream>>>(
        xg, gw, eis, eis + Ecnt, agg, Ecnt, p.K, cout/4, cout);
    k_finalize<<<CDIVL((long long)n * cout, 256), 256, 0, stream>>>(
        agg, degv, rootout, p.bias, out, n, cout);
  };
  auto runBn = [&](const float* in, float* out, int n, int C, const Bn& p,
                   int mode, const float* skip) {
    k_bn_stats<<<C, 256, 0, stream>>>(in, n, C, stats);
    k_bn_apply<<<CDIVL((long long)n * C, 256), 256, 0, stream>>>(
        in, stats, p.gamma, p.beta, skip, out, n, C, mode);
  };
  auto pool = [&](const float* src, const int* cluster, int n_in2, float* out,
                  int n_out, int C) {
    hipMemsetAsync(umax, 0, (size_t)n_out * C * 4, stream);
    k_pool_scatter<<<CDIVL((long long)n_in2 * C, 256), 256, 0, stream>>>(
        src, cluster, umax, n_in2, C);
    k_pool_decode<<<CDIVL((long long)n_out * C, 256), 256, 0, stream>>>(
        umax, out, n_out, C);
  };
  auto runRes = [&](const float* xin, int n, int cin, int cout, const Res& r,
                    const int* eis, int Ecnt, const float* eaf,
                    const float* degv, float* out) {
    runConv(xin, n, cin,  cout, r.lc1, eis, Ecnt, eaf, degv, th1);
    runBn(th1, th1, n, cout, r.lbn1, 1, nullptr);                 // elu(bn(.))
    runConv(th1, n, cout, cout, r.lc2, eis, Ecnt, eaf, degv, th2);
    runConv(xin, n, cin,  cout, r.sc,  eis, Ecnt, eaf, degv, ts);
    runBn(ts,  ts,  n, cout, r.sbn,  0, nullptr);                 // bn shortcut
    runBn(th2, out, n, cout, r.lbn2, 2, ts);                      // elu(bn(h)+s)
  };

  // ---- degrees (depend only on graph structure per level) ----
  for (int l = 0; l < 4; ++l) {
    hipMemsetAsync(degB[l], 0, (size_t)nodes[l] * 4, stream);
    k_edge_deg<<<CDIVL(E[l], 256), 256, 0, stream>>>(ei[l] + E[l], degB[l], E[l]);
  }

  // ---- pipeline ----
  runConv(x, N, 3, 24, conv1, ei[0], E[0], ea[0], degB[0], h0);
  runBn(h0, h0, N, 24, bn1, 1, nullptr);
  pool(h0, cl[0], N, p1, n1, 24);

  runRes(p1, n1, 24, 48,  RB[0], ei[1], E[1], ea[1], degB[1], b1o);
  pool(b1o, cl[1], n1, p2, n2, 48);

  runRes(p2, n2, 48, 96,  RB[1], ei[2], E[2], ea[2], degB[2], b2o);
  pool(b2o, cl[2], n2, p3, n3, 96);

  runRes(p3, n3, 96, 192, RB[2], ei[3], E[3], ea[3], degB[3], b3o);
  pool(b3o, cl[3], n3, p4, n4, 192);

  // global mean pool -> d_out (G*192 floats, viewed as [8,5,192])
  hipMemsetAsync(gsum, 0, (size_t)G_GRAPHS * 192 * 4, stream);
  hipMemsetAsync(gcnt, 0, (size_t)G_GRAPHS * 4, stream);
  k_graph_sum<<<CDIVL((long long)n4 * 192, 256), 256, 0, stream>>>(
      p4, batch4, gsum, gcnt, n4, 192);
  k_graph_mean<<<CDIVL((long long)G_GRAPHS * 192, 256), 256, 0, stream>>>(
      gsum, gcnt, (float*)d_out, 192);
  (void)n_in; (void)out_size; (void)ws_size;
}
